// AttentionGCN_49323404427606
// MI455X (gfx1250) — compile-verified, matrix operand order checked by reading, not run on hardware
//
#include <hip/hip_runtime.h>
#include <hip/hip_bf16.h>
#include <stdint.h>

#define BB      2048
#define SS      10
#define DD      128
#define HH      128
#define HFP     136    // f16 row pitch (128 + 8 pad) -> 272B row, kills LDS bank conflicts
#define GATES   512    // 4*HH
#define NHEADS  8
#define NEG_SLOPE 0.1f

typedef __attribute__((ext_vector_type(16))) _Float16 v16h;
typedef __attribute__((ext_vector_type(8)))  float    v8f;

union V16H { v16h v; uint32_t u[8]; };

// ---------------------------------------------------------------------------
// Gather feature rows (128 wide) by index; optional f32 and f16 outputs.
// ---------------------------------------------------------------------------
__global__ __launch_bounds__(128) void gather_rows_kernel(
    const float* __restrict__ raw, const int* __restrict__ idx,
    float* __restrict__ of32, _Float16* __restrict__ of16)
{
    const int row = blockIdx.x;
    const int d   = threadIdx.x;
    const int id  = idx[row];
    const float v = raw[(size_t)id * DD + d];
    if (of32) of32[(size_t)row * DD + d] = v;
    if (of16) of16[(size_t)row * DD + d] = (_Float16)v;
}

// gather with concatenated index sources: rows [0,nA) from idxA, rest from idxB
__global__ __launch_bounds__(128) void gather_rows2_kernel(
    const float* __restrict__ raw, const int* __restrict__ idxA, int nA,
    const int* __restrict__ idxB, float* __restrict__ of32)
{
    const int row = blockIdx.x;
    const int d   = threadIdx.x;
    const int id  = (row < nA) ? idxA[row] : idxB[row - nA];
    of32[(size_t)row * DD + d] = raw[(size_t)id * DD + d];
}

__global__ void f32_to_f16_kernel(const float* __restrict__ in,
                                  _Float16* __restrict__ out, int n)
{
    const int i = blockIdx.x * blockDim.x + threadIdx.x;
    if (i < n) out[i] = (_Float16)in[i];
}

// ---------------------------------------------------------------------------
// Fused bidirectional-capable LSTM direction kernel.
//   One 256-thread block (8 waves) owns a 16-node tile for all S steps.
//   - Each wave pins its 4 gate-tiles of Wih and Whh in VGPRs (256 VGPRs)
//     for the whole kernel (SSA values -> immune to asm memory clobbers).
//   - Per step the 16x128 f16 x_s tile is staged into LDS with one
//     global_load_async_to_lds_b128 per thread (ASYNCcnt path), and the
//     next step's tile is prefetched into near caches.
//   - Inner loop: one A chunk (2x ds_load_b128) feeds 4 WMMAs into 4
//     persistent f32 accumulators (gates = x_s @ Wih^T + h @ Whh^T).
//   - 256 threads then apply the LSTM gate nonlinearities.
// ---------------------------------------------------------------------------
__global__ __launch_bounds__(256) void lstm_dir_kernel(
    const _Float16* __restrict__ Xf16,   // [N, SS, DD] f16
    const _Float16* __restrict__ Wih,    // [GATES, DD] f16 row-major
    const _Float16* __restrict__ Whh,    // [GATES, HH] f16 row-major
    const float* __restrict__ bih, const float* __restrict__ bhh,
    float* __restrict__ outH,            // [N, SS, HH] f32 hidden states
    int reverse)
{
    __shared__ float    s_gates[16 * GATES]; // 32 KB
    __shared__ float    s_c[16 * HH];        //  8 KB
    __shared__ _Float16 s_hf[16 * HFP];      //  4.25 KB (h in f16, padded rows)
    __shared__ _Float16 s_x[16 * HFP];       //  4.25 KB (x_s staging, padded rows)

    const int tid   = threadIdx.x;
    const int wave  = tid >> 5;
    const int lane  = tid & 31;
    const int lh    = lane >> 4;   // lane half: selects K=8..15 / 24..31 group
    const int r     = lane & 15;   // row (A) / column (B) within tile
    const int node0 = blockIdx.x * 16;

    // K offsets for packed-pair loads per the ISA 16-bit A/B VGPR layout
    int koff[8];
#pragma unroll
    for (int j = 0; j < 8; ++j)
        koff[j] = 8 * lh + (j < 4 ? 2 * j : 16 + 2 * (j - 4));

    // ---- pin this wave's weight tiles in VGPRs for the whole kernel ----
    V16H wih[4][4], whh[4][4];   // [gate tile][K chunk], 8 VGPRs each = 256 VGPRs
#pragma unroll
    for (int t = 0; t < 4; ++t) {
        const int g0 = (wave * 4 + t) * 16;
        const _Float16* BrowI = Wih + (size_t)(g0 + r) * DD;
        const _Float16* BrowH = Whh + (size_t)(g0 + r) * HH;
#pragma unroll
        for (int c = 0; c < 4; ++c) {
#pragma unroll
            for (int j = 0; j < 8; ++j) {
                wih[t][c].u[j] = *(const uint32_t*)(BrowI + c * 32 + koff[j]);
                whh[t][c].u[j] = *(const uint32_t*)(BrowH + c * 32 + koff[j]);
            }
        }
    }

    for (int i = tid; i < 16 * HH; i += 256) s_c[i] = 0.f;
    for (int i = tid; i < 16 * HFP; i += 256) s_hf[i] = (_Float16)0.f;
    __syncthreads();

    // per-thread slice of the 16x128 x-tile: 16 bytes each (256*16B = 4KB)
    const int xm = tid >> 4;           // node row 0..15
    const int xc = (tid & 15) * 8;     // f16 column 0..120 step 8
    const uint32_t x_ldsoff =
        (uint32_t)(uintptr_t)s_x + (uint32_t)(xm * HFP + xc) * 2;

    for (int ss = 0; ss < SS; ++ss) {
        const int s = reverse ? (SS - 1 - ss) : ss;

        // ---- stage x_s tile into LDS via async copy (ASYNCcnt) ----
        {
            const _Float16* gsrc =
                Xf16 + ((size_t)(node0 + xm) * SS + s) * DD + xc;
            asm volatile("global_load_async_to_lds_b128 %0, %1, off"
                         :: "v"(x_ldsoff), "v"((uint64_t)(uintptr_t)gsrc)
                         : "memory");
            // warm near caches for the next step's tile
            const int sn = reverse ? (s - 1) : (s + 1);
            if (0 <= sn && sn < SS)
                __builtin_prefetch(
                    Xf16 + ((size_t)(node0 + xm) * SS + sn) * DD + xc, 0, 3);
            asm volatile("s_wait_asynccnt 0" ::: "memory");
        }
        __syncthreads();

        // ---- gates = x_s @ Wih^T + h @ Whh^T (WMMA, f32 accum) ----
        v8f acc[4] = {{}, {}, {}, {}};

        const _Float16* Ax = s_x + r * HFP;
#pragma unroll
        for (int c = 0; c < 4; ++c) {
            V16H a;
#pragma unroll
            for (int j = 0; j < 8; ++j)
                a.u[j] = *(const uint32_t*)(Ax + c * 32 + koff[j]);
#pragma unroll
            for (int t = 0; t < 4; ++t)
                acc[t] = __builtin_amdgcn_wmma_f32_16x16x32_f16(
                           false, a.v, false, wih[t][c].v, (short)0, acc[t],
                           false, false);
        }

        const _Float16* Ah = s_hf + r * HFP;
#pragma unroll
        for (int c = 0; c < 4; ++c) {
            V16H a;
#pragma unroll
            for (int j = 0; j < 8; ++j)
                a.u[j] = *(const uint32_t*)(Ah + c * 32 + koff[j]);
#pragma unroll
            for (int t = 0; t < 4; ++t)
                acc[t] = __builtin_amdgcn_wmma_f32_16x16x32_f16(
                           false, a.v, false, whh[t][c].v, (short)0, acc[t],
                           false, false);
        }

        // C layout: lane -> col N = r, rows M = i + 8*lh
#pragma unroll
        for (int t = 0; t < 4; ++t) {
            const int g0 = (wave * 4 + t) * 16;
#pragma unroll
            for (int i = 0; i < 8; ++i)
                s_gates[(size_t)(i + 8 * lh) * GATES + g0 + r] = acc[t][i];
        }
        __syncthreads();

        // ---- LSTM gate nonlinearities: 16 nodes x 128 hidden ----
        for (int e = tid; e < 16 * HH; e += 256) {
            const int m = e >> 7, j = e & 127;
            const float* g = s_gates + (size_t)m * GATES;
            const float gi = g[j]          + bih[j]          + bhh[j];
            const float gf = g[HH + j]     + bih[HH + j]     + bhh[HH + j];
            const float gg = g[2 * HH + j] + bih[2 * HH + j] + bhh[2 * HH + j];
            const float go = g[3 * HH + j] + bih[3 * HH + j] + bhh[3 * HH + j];
            const float iv = 1.f / (1.f + expf(-gi));
            const float fv = 1.f / (1.f + expf(-gf));
            const float gv = tanhf(gg);
            const float ov = 1.f / (1.f + expf(-go));
            const float c  = fv * s_c[e] + iv * gv;
            const float h  = ov * tanhf(c);
            s_c[e] = c;
            s_hf[m * HFP + j] = (_Float16)h;
            outH[((size_t)(node0 + m) * SS + s) * HH + j] = h;
        }
        __syncthreads();
    }
}

// ---------------------------------------------------------------------------
// Attention + aggregation per node.
//   att[h,s] = softmax_s(leaky_relu(Wm[h] . [out_f|out_b|x]))
//   out      = sum_s (mean_h att[h,s]) * neighbor_x[s]
// ---------------------------------------------------------------------------
__global__ __launch_bounds__(128) void att_agg_kernel(
    const float* __restrict__ outF, const float* __restrict__ outB, // [N,SS,HH]
    const float* __restrict__ x,    // [N,DD]
    const float* __restrict__ Wm,   // [NHEADS, DD + 2*HH]
    const float* __restrict__ nx,   // [N,SS,DD]
    float* __restrict__ out)        // [N,DD]
{
    __shared__ float s_att[NHEADS][SS];
    __shared__ float s_w[SS];
    const int n   = blockIdx.x;
    const int tid = threadIdx.x;

    if (tid < NHEADS * SS) {
        const int h = tid / SS, s = tid % SS;
        const float* wm = Wm + (size_t)h * (DD + 2 * HH);
        const float* of = outF + ((size_t)n * SS + s) * HH;
        const float* ob = outB + ((size_t)n * SS + s) * HH;
        const float* xr = x + (size_t)n * DD;
        float acc = 0.f;
        for (int k = 0; k < HH; ++k) acc += wm[k] * of[k];
        for (int k = 0; k < HH; ++k) acc += wm[HH + k] * ob[k];
        for (int k = 0; k < DD; ++k) acc += wm[2 * HH + k] * xr[k];
        s_att[h][s] = (acc > 0.f) ? acc : NEG_SLOPE * acc;
    }
    __syncthreads();

    if (tid < NHEADS) {
        float mx = -3.4e38f;
        for (int s = 0; s < SS; ++s) mx = fmaxf(mx, s_att[tid][s]);
        float e[SS], sum = 0.f;
        for (int s = 0; s < SS; ++s) { e[s] = expf(s_att[tid][s] - mx); sum += e[s]; }
        const float inv = 1.f / sum;
        for (int s = 0; s < SS; ++s) s_att[tid][s] = e[s] * inv;
    }
    __syncthreads();

    if (tid < SS) {
        float w = 0.f;
        for (int h = 0; h < NHEADS; ++h) w += s_att[h][tid];
        s_w[tid] = w * (1.f / NHEADS);
    }
    __syncthreads();

    float acc = 0.f;
    for (int s = 0; s < SS; ++s)
        acc += s_w[s] * nx[((size_t)n * SS + s) * DD + tid];
    out[(size_t)n * DD + tid] = acc;
}

// ---------------------------------------------------------------------------
extern "C" void kernel_launch(void* const* d_in, const int* in_sizes, int n_in,
                              void* d_out, int out_size, void* d_ws, size_t ws_size,
                              hipStream_t stream)
{
    (void)in_sizes; (void)n_in; (void)out_size; (void)ws_size;

    const float* raw    = (const float*)d_in[0];
    const float* Wih_f1 = (const float*)d_in[1];
    const float* Whh_f1 = (const float*)d_in[2];
    const float* bih_f1 = (const float*)d_in[3];
    const float* bhh_f1 = (const float*)d_in[4];
    const float* Wih_b1 = (const float*)d_in[5];
    const float* Whh_b1 = (const float*)d_in[6];
    const float* bih_b1 = (const float*)d_in[7];
    const float* bhh_b1 = (const float*)d_in[8];
    const float* Wm1    = (const float*)d_in[9];
    const float* Wih_f2 = (const float*)d_in[10];
    const float* Whh_f2 = (const float*)d_in[11];
    const float* bih_f2 = (const float*)d_in[12];
    const float* bhh_f2 = (const float*)d_in[13];
    const float* Wih_b2 = (const float*)d_in[14];
    const float* Whh_b2 = (const float*)d_in[15];
    const float* bih_b2 = (const float*)d_in[16];
    const float* bhh_b2 = (const float*)d_in[17];
    const float* Wm2    = (const float*)d_in[18];
    const int* batch_ids = (const int*)d_in[19];
    const int* neigh1    = (const int*)d_in[20];
    const int* neigh_l1  = (const int*)d_in[21];

    const int N1 = BB * (SS + 1);   // 22528 layer-1 nodes (multiple of 16)
    const int R1 = N1 * SS;         // 225280 layer-1 neighbor rows
    const int N2 = BB;              // 2048  layer-2 nodes (multiple of 16)
    const int R2 = N2 * SS;         // 20480 layer-2 neighbor rows

    char* ws = (char*)d_ws;
    size_t off = 0;
    auto carve = [&](size_t bytes) -> char* {
        char* p = ws + off;
        off += (bytes + 255) & ~(size_t)255;
        return p;
    };

    _Float16* Xn1_16 = (_Float16*)carve((size_t)R1 * DD * 2);
    float*    Xn1_32 = (float*)   carve((size_t)R1 * DD * 4);
    float*    x1_32  = (float*)   carve((size_t)N1 * DD * 4);
    float*    outF1  = (float*)   carve((size_t)R1 * HH * 4);
    float*    outB1  = (float*)   carve((size_t)R1 * HH * 4);
    float*    h1     = (float*)   carve((size_t)N1 * DD * 4);
    _Float16* Xn2_16 = (_Float16*)carve((size_t)R2 * DD * 2);
    float*    outF2  = (float*)   carve((size_t)R2 * HH * 4);
    float*    outB2  = (float*)   carve((size_t)R2 * HH * 4);

    const float* w32[8] = {Wih_f1, Whh_f1, Wih_b1, Whh_b1,
                           Wih_f2, Whh_f2, Wih_b2, Whh_b2};
    _Float16* w16[8];
    for (int i = 0; i < 8; ++i) w16[i] = (_Float16*)carve((size_t)GATES * DD * 2);

    // weights -> f16
    for (int i = 0; i < 8; ++i)
        f32_to_f16_kernel<<<(GATES * DD + 255) / 256, 256, 0, stream>>>(
            w32[i], w16[i], GATES * DD);

    // layer-1 gathers
    gather_rows_kernel<<<R1, 128, 0, stream>>>(raw, neigh_l1, Xn1_32, Xn1_16);
    gather_rows2_kernel<<<N1, 128, 0, stream>>>(raw, batch_ids, BB, neigh1, x1_32);

    // layer-1 bi-LSTM
    lstm_dir_kernel<<<N1 / 16, 256, 0, stream>>>(Xn1_16, w16[0], w16[1],
                                                 bih_f1, bhh_f1, outF1, 0);
    lstm_dir_kernel<<<N1 / 16, 256, 0, stream>>>(Xn1_16, w16[2], w16[3],
                                                 bih_b1, bhh_b1, outB1, 1);

    // layer-1 attention + aggregation -> h1
    att_agg_kernel<<<N1, 128, 0, stream>>>(outF1, outB1, x1_32, Wm1, Xn1_32, h1);

    // layer-2: neighbor features = h1[B:], self = h1[:B]
    f32_to_f16_kernel<<<(R2 * DD + 255) / 256, 256, 0, stream>>>(
        h1 + (size_t)BB * DD, Xn2_16, R2 * DD);
    lstm_dir_kernel<<<N2 / 16, 256, 0, stream>>>(Xn2_16, w16[4], w16[5],
                                                 bih_f2, bhh_f2, outF2, 0);
    lstm_dir_kernel<<<N2 / 16, 256, 0, stream>>>(Xn2_16, w16[6], w16[7],
                                                 bih_b2, bhh_b2, outB2, 1);
    att_agg_kernel<<<N2, 128, 0, stream>>>(outF2, outB2, h1, Wm2,
                                           h1 + (size_t)BB * DD, (float*)d_out);
}
